// GPTBigCodeAttention_56856777064683
// MI455X (gfx1250) — matrix operand order repaired and checked by reading, not compiled, not run on hardware
//
#include <hip/hip_runtime.h>

// ---------------------------------------------------------------------------
// GPTBigCode multi-query attention, MI455X (gfx1250, wave32, WMMA bf16)
// B=2, S=2048, D=2048, H=16, hd=128
//  - GEMMs: LDS-staged, double-buffered via GLOBAL_LOAD_ASYNC_TO_LDS_B128
//    (ASYNCcnt pipeline), operands read as ds_load_b128 pairs, bf16 WMMA.
//  - Attention: flash MQA, WMMA bf16, V stored transposed so all operands
//    are contiguous 16B loads.
// ---------------------------------------------------------------------------

typedef __bf16 bf16;
typedef __attribute__((ext_vector_type(16))) __bf16 v16bf;
typedef __attribute__((ext_vector_type(8)))  __bf16 bf16x8;
typedef __attribute__((ext_vector_type(8)))  float  v8f;

static __device__ __forceinline__ v8f wmma_bf16(v16bf a, v16bf b, v8f c) {
    return __builtin_amdgcn_wmma_f32_16x16x32_bf16(false, a, false, b, (short)0, c,
                                                   false, false);
}

static __device__ __forceinline__ v16bf load_op16(const bf16* p0, const bf16* p1) {
    bf16x8 lo = *(const bf16x8*)p0;
    bf16x8 hi = *(const bf16x8*)p1;
    v16bf r;
#pragma unroll
    for (int e = 0; e < 8; ++e) { r[e] = lo[e]; r[8 + e] = hi[e]; }
    return r;
}

// Async 16B global -> LDS copy (tracked by ASYNCcnt, data lands in LDS).
static __device__ __forceinline__ void async_load_b128(unsigned lds_addr,
                                                       const void* gaddr) {
    asm volatile("global_load_async_to_lds_b128 %0, %1, off"
                 :
                 : "v"(lds_addr), "v"(gaddr)
                 : "memory");
}
static __device__ __forceinline__ void wait_async_le(int n) {
    if (n == 0)
        asm volatile("s_wait_asynccnt 0" ::: "memory");
    else
        asm volatile("s_wait_asynccnt 4" ::: "memory");
}

// ---------------------------------------------------------------------------
// f32 -> bf16 elementwise convert
// ---------------------------------------------------------------------------
__global__ void cvt_f32_bf16(const float* __restrict__ in, bf16* __restrict__ out,
                             size_t n) {
    size_t i = (size_t)blockIdx.x * blockDim.x + threadIdx.x;
    if (i < n) out[i] = (bf16)in[i];
}

// ---------------------------------------------------------------------------
// f32 [R][C] -> bf16 [C][R] tiled transpose-convert
// ---------------------------------------------------------------------------
__global__ __launch_bounds__(256) void cvt_transpose_f32_bf16(
    const float* __restrict__ in, bf16* __restrict__ out, int R, int C) {
    __shared__ float tile[32][33];
    const int c0 = blockIdx.x * 32;
    const int r0 = blockIdx.y * 32;
    const int tx = threadIdx.x;
    const int ty = threadIdx.y;
#pragma unroll
    for (int i = 0; i < 32; i += 8)
        tile[ty + i][tx] = in[(size_t)(r0 + ty + i) * C + (c0 + tx)];
    __syncthreads();
#pragma unroll
    for (int i = 0; i < 32; i += 8)
        out[(size_t)(c0 + ty + i) * R + (r0 + tx)] = (bf16)tile[tx][ty + i];
}

// ---------------------------------------------------------------------------
// GEMM: C[M,N] = A[M,K] @ W[K,N] + bias, W supplied transposed (Wt[N][K]).
// Block 256 threads (8 waves); block tile 128(M) x 128(N); wave tile 64x32.
// A/B k-slabs (128x32 bf16 each) staged in LDS, double-buffered with
// GLOBAL_LOAD_ASYNC_TO_LDS_B128; rows padded to 40 elems (80B) for banking.
// mode 0: split-store QKV (Q[row][c], K[row][c-2048], Vt[b][c-2176][s])
// mode 1: f32 store to fout
// ---------------------------------------------------------------------------
#define PADROW 40  // bf16 elements per LDS row (80 bytes, 16B aligned)

__global__ __launch_bounds__(256) void gemm_bf16(
    const bf16* __restrict__ A, const bf16* __restrict__ Wt,
    const float* __restrict__ bias, int M, int N, int K, int mode,
    bf16* __restrict__ qout, bf16* __restrict__ kout, bf16* __restrict__ vtout,
    float* __restrict__ fout) {
    __shared__ bf16 smem[2][2][128 * PADROW];  // [buf][A|B][row*PADROW+k] 40KB

    const int lane = threadIdx.x & 31;
    const int wave = threadIdx.x >> 5;
    const int wm = wave & 1;   // 2 over M
    const int wn = wave >> 1;  // 4 over N
    const int mblk = blockIdx.x * 128;
    const int nblk = blockIdx.y * 128;
    const int half = lane >> 4;
    const int l16 = lane & 15;

    const unsigned ldsBase = (unsigned)(uintptr_t)(&smem[0][0][0]);
    const unsigned bufStride = 2u * 128u * PADROW * 2u;  // bytes per buffer
    const unsigned matStride = 128u * PADROW * 2u;       // bytes per matrix

    // Issue one k-slab stage: 128x32 of A and of Wt, 16B chunks, 4 per thread.
    auto issue_stage = [&](int k0, int buf) {
        const int t = (int)threadIdx.x;
#pragma unroll
        for (int i = 0; i < 2; ++i) {
            const int c = t + 256 * i;     // 0..511
            const int row = c >> 2;        // 0..127
            const int kc = (c & 3) * 8;    // 0,8,16,24
            const unsigned lo = (unsigned)(row * PADROW + kc) * 2u;
            async_load_b128(ldsBase + (unsigned)buf * bufStride + lo,
                            A + (size_t)(mblk + row) * K + k0 + kc);
            async_load_b128(ldsBase + (unsigned)buf * bufStride + matStride + lo,
                            Wt + (size_t)(nblk + row) * K + k0 + kc);
        }
    };

    v8f acc[4][2];
#pragma unroll
    for (int t = 0; t < 4; ++t)
#pragma unroll
        for (int nt = 0; nt < 2; ++nt)
#pragma unroll
            for (int r = 0; r < 8; ++r) acc[t][nt][r] = 0.f;

    const int nk = K / 32;
    issue_stage(0, 0);
    for (int i = 0; i < nk; ++i) {
        const int buf = i & 1;
        if (i + 1 < nk) {
            issue_stage((i + 1) * 32, buf ^ 1);
            wait_async_le(4);  // stage i's 4 async loads (per wave) are done
        } else {
            wait_async_le(0);
        }
        __syncthreads();  // all waves' slab-i data visible in LDS

        const bf16* AL = &smem[buf][0][0];
        const bf16* BL = &smem[buf][1][0];
        v16bf bb[2];
#pragma unroll
        for (int nt = 0; nt < 2; ++nt) {
            const bf16* bp = BL + (wn * 32 + nt * 16 + l16) * PADROW + half * 8;
            bb[nt] = load_op16(bp, bp + 16);
        }
#pragma unroll
        for (int t = 0; t < 4; ++t) {
            const bf16* ap = AL + (wm * 64 + t * 16 + l16) * PADROW + half * 8;
            v16bf amat = load_op16(ap, ap + 16);
#pragma unroll
            for (int nt = 0; nt < 2; ++nt)
                acc[t][nt] = wmma_bf16(amat, bb[nt], acc[t][nt]);
        }
        __syncthreads();  // protect buf from stage i+2's async writes
    }

    // Epilogue. acc[t][nt][r] -> row mblk+wm*64+t*16+half*8+r, col nblk+wn*32+nt*16+l16
#pragma unroll
    for (int nt = 0; nt < 2; ++nt) {
        const int col = nblk + wn * 32 + nt * 16 + l16;
        const float bval = bias[col];
#pragma unroll
        for (int t = 0; t < 4; ++t) {
#pragma unroll
            for (int r = 0; r < 8; ++r) {
                const int row = mblk + wm * 64 + t * 16 + half * 8 + r;
                const float v = acc[t][nt][r] + bval;
                if (mode == 0) {
                    if (col < 2048) {
                        qout[(size_t)row * 2048 + col] = (bf16)v;
                    } else if (col < 2176) {
                        kout[(size_t)row * 128 + (col - 2048)] = (bf16)v;
                    } else {
                        const int bb_ = row >> 11;
                        const int ss = row & 2047;
                        vtout[((size_t)bb_ * 128 + (col - 2176)) * 2048 + ss] = (bf16)v;
                    }
                } else {
                    fout[(size_t)row * N + col] = v;
                }
            }
        }
    }
}

// ---------------------------------------------------------------------------
// Flash multi-query attention. One wave per 16-query tile per (b,h).
// Q:[B*S,2048] bf16; Kc:[B*S,128] bf16; Vt:[B][128][S] bf16; O:[B*S,2048] bf16
// ---------------------------------------------------------------------------
__global__ __launch_bounds__(128) void mqa_attn(
    const bf16* __restrict__ Q, const bf16* __restrict__ Kc,
    const bf16* __restrict__ Vt, bf16* __restrict__ O, int S) {
    __shared__ bf16 pbuf[4][16 * 32];
    const int wave = threadIdx.x >> 5;
    const int lane = threadIdx.x & 31;
    const int half = lane >> 4;
    const int l16 = lane & 15;

    const int tileIdx = blockIdx.x * 4 + wave;
    const int qtPerHead = S / 16;
    const int qt = tileIdx % qtPerHead;
    const int h = (tileIdx / qtPerHead) % 16;
    const int b = tileIdx / (qtPerHead * 16);
    const int q0 = qt * 16;

    const bf16* Qb = Q + (size_t)b * S * 2048 + h * 128;
    const bf16* Kb = Kc + (size_t)b * S * 128;
    const bf16* Vb = Vt + (size_t)b * 128 * S;

    v16bf qa[4];
#pragma unroll
    for (int c = 0; c < 4; ++c) {
        const bf16* qp = Qb + (size_t)(q0 + l16) * 2048 + c * 32 + half * 8;
        qa[c] = load_op16(qp, qp + 16);
    }

    v8f oacc[8];
#pragma unroll
    for (int dt = 0; dt < 8; ++dt)
#pragma unroll
        for (int r = 0; r < 8; ++r) oacc[dt][r] = 0.f;
    float mrow[8], lrow[8];
#pragma unroll
    for (int r = 0; r < 8; ++r) { mrow[r] = -3.0e38f; lrow[r] = 0.f; }

    const float scale = 0.08838834764831845f;  // 1/sqrt(128)
    const int kend = q0 + 16;

    for (int k0 = 0; k0 < kend; k0 += 32) {
        if (k0 + 32 < kend)
            __builtin_prefetch(Kb + (size_t)(k0 + 32 + lane) * 128, 0, 1);

        v8f sc[2];
#pragma unroll
        for (int kt = 0; kt < 2; ++kt) {
#pragma unroll
            for (int r = 0; r < 8; ++r) sc[kt][r] = 0.f;
            const int key = k0 + kt * 16 + l16;
#pragma unroll
            for (int c = 0; c < 4; ++c) {
                const bf16* kp = Kb + (size_t)key * 128 + c * 32 + half * 8;
                sc[kt] = wmma_bf16(qa[c], load_op16(kp, kp + 16), sc[kt]);
            }
        }

        float pv0[8], pv1[8];
#pragma unroll
        for (int r = 0; r < 8; ++r) {
            const int qrow = q0 + half * 8 + r;
            float s0 = sc[0][r] * scale;
            float s1 = sc[1][r] * scale;
            if (k0 + l16 > qrow) s0 = -3.0e38f;
            if (k0 + 16 + l16 > qrow) s1 = -3.0e38f;
            float mx = fmaxf(s0, s1);
#pragma unroll
            for (int off = 1; off < 16; off <<= 1)
                mx = fmaxf(mx, __shfl_xor(mx, off, 32));
            const float mnew = fmaxf(mrow[r], mx);
            const float corr = __expf(mrow[r] - mnew);
            const float p0 = __expf(s0 - mnew);
            const float p1 = __expf(s1 - mnew);
            float rs = p0 + p1;
#pragma unroll
            for (int off = 1; off < 16; off <<= 1) rs += __shfl_xor(rs, off, 32);
            lrow[r] = lrow[r] * corr + rs;
            mrow[r] = mnew;
#pragma unroll
            for (int dt = 0; dt < 8; ++dt) oacc[dt][r] *= corr;
            pv0[r] = p0;
            pv1[r] = p1;
        }

        bf16* pb = pbuf[wave];
#pragma unroll
        for (int r = 0; r < 8; ++r) {
            pb[(half * 8 + r) * 32 + l16] = (bf16)pv0[r];
            pb[(half * 8 + r) * 32 + 16 + l16] = (bf16)pv1[r];
        }
        __asm__ volatile("s_wait_dscnt 0" ::: "memory");
        v16bf pa;
        {
            const bf16* pr = pbuf[wave] + l16 * 32;
#pragma unroll
            for (int e = 0; e < 8; ++e) {
                pa[e] = pr[half * 8 + e];
                pa[8 + e] = pr[16 + half * 8 + e];
            }
        }

#pragma unroll
        for (int dt = 0; dt < 8; ++dt) {
            const bf16* vp = Vb + (size_t)(dt * 16 + l16) * S + k0 + half * 8;
            oacc[dt] = wmma_bf16(pa, load_op16(vp, vp + 16), oacc[dt]);
        }
    }

    bf16* Ob = O + (size_t)b * S * 2048 + h * 128;
#pragma unroll
    for (int r = 0; r < 8; ++r) {
        const float inv = 1.f / lrow[r];
        const int row = q0 + half * 8 + r;
#pragma unroll
        for (int dt = 0; dt < 8; ++dt)
            Ob[(size_t)row * 2048 + dt * 16 + l16] = (bf16)(oacc[dt][r] * inv);
    }
}

// ---------------------------------------------------------------------------
// Host-side launch
// ---------------------------------------------------------------------------
extern "C" void kernel_launch(void* const* d_in, const int* in_sizes, int n_in,
                              void* d_out, int out_size, void* d_ws, size_t ws_size,
                              hipStream_t stream) {
    (void)in_sizes; (void)n_in; (void)out_size; (void)ws_size;
    const int B = 2, S = 2048, D = 2048, H = 16, HD = 128;
    const int M = B * S;          // 4096
    const int Nqkv = D + 2 * HD;  // 2304

    const float* hs = (const float*)d_in[0];
    const float* c_attn_w = (const float*)d_in[1];
    const float* c_attn_b = (const float*)d_in[2];
    const float* c_proj_w = (const float*)d_in[3];
    const float* c_proj_b = (const float*)d_in[4];
    float* out = (float*)d_out;

    char* ws = (char*)d_ws;
    bf16* Xbf = (bf16*)ws;     ws += (size_t)M * D * 2;
    bf16* WqkvT = (bf16*)ws;   ws += (size_t)D * Nqkv * 2;
    bf16* WprojT = (bf16*)ws;  ws += (size_t)D * D * 2;
    bf16* Qbf = (bf16*)ws;     ws += (size_t)M * D * 2;
    bf16* Kbf = (bf16*)ws;     ws += (size_t)M * HD * 2;
    bf16* VtBf = (bf16*)ws;    ws += (size_t)M * HD * 2;
    bf16* Abf = (bf16*)ws;     ws += (size_t)M * D * 2;

    {
        size_t n1 = (size_t)M * D;
        cvt_f32_bf16<<<(unsigned)((n1 + 255) / 256), 256, 0, stream>>>(hs, Xbf, n1);
        dim3 blk(32, 8);
        dim3 g1(Nqkv / 32, D / 32);
        cvt_transpose_f32_bf16<<<g1, blk, 0, stream>>>(c_attn_w, WqkvT, D, Nqkv);
        dim3 g2(D / 32, D / 32);
        cvt_transpose_f32_bf16<<<g2, blk, 0, stream>>>(c_proj_w, WprojT, D, D);
    }
    {
        dim3 grid(M / 128, Nqkv / 128);
        gemm_bf16<<<grid, 256, 0, stream>>>(Xbf, WqkvT, c_attn_b, M, Nqkv, D,
                                            0, Qbf, Kbf, VtBf, nullptr);
    }
    {
        const int tiles = B * H * (S / 16);
        mqa_attn<<<tiles / 4, 128, 0, stream>>>(Qbf, Kbf, VtBf, Abf, S);
    }
    {
        dim3 grid(M / 128, D / 128);
        gemm_bf16<<<grid, 256, 0, stream>>>(Abf, WprojT, c_proj_b, M, D, D,
                                            1, nullptr, nullptr, nullptr, out);
    }
}
